// Int8Linear_28810640621892
// MI455X (gfx1250) — 8-bit matrix operand order repaired and checked by reading, compiled, not run on hardware
//
#include <hip/hip_runtime.h>
#include <stdint.h>

typedef __attribute__((ext_vector_type(8))) int v8i;

#define IN_F  4096
#define OUT_F 4096
#define BM 128
#define BN 128
#define BK 64
#define LDT 80
#define TILE_B (BM * LDT)
#define NT (IN_F / BK)
#define NBUF 4

static __device__ __forceinline__ int clampi(int v, int lo, int hi) {
    return v < lo ? lo : (v > hi ? hi : v);
}

__global__ __launch_bounds__(256) void Int8Linear_repack_w(const int* __restrict__ w,
                                                           uint32_t* __restrict__ w8) {
    const int i = blockIdx.x * 256 + threadIdx.x;
    const int4 v = reinterpret_cast<const int4*>(w)[i];
    const uint32_t p = (uint32_t)(v.x & 0xff)         |
                       ((uint32_t)(v.y & 0xff) << 8)  |
                       ((uint32_t)(v.z & 0xff) << 16) |
                       ((uint32_t)(v.w & 0xff) << 24);
    w8[i] = p;
}

__global__ __launch_bounds__(256) void Int8Linear_quantize(const float* __restrict__ x,
                                                           uint32_t* __restrict__ qx,
                                                           float* __restrict__ xs) {
    const int lane  = threadIdx.x & 31;
    const int wave  = threadIdx.x >> 5;
    const int token = blockIdx.x * 8 + wave;
    const float4* row = reinterpret_cast<const float4*>(x + (size_t)token * IN_F);

    float amax = 0.0f;
#pragma unroll 4
    for (int i = 0; i < 32; ++i) {
        const float4 v = row[i * 32 + lane];
        amax = fmaxf(amax, fmaxf(fmaxf(fabsf(v.x), fabsf(v.y)),
                                 fmaxf(fabsf(v.z), fabsf(v.w))));
    }
#pragma unroll
    for (int off = 16; off > 0; off >>= 1)
        amax = fmaxf(amax, __shfl_xor(amax, off, 32));

    const float scale = amax * (1.0f / 127.0f);
    const float inv   = (amax > 0.0f) ? (127.0f / amax) : 0.0f;
    if (lane == 0) xs[token] = scale;

#pragma unroll 4
    for (int i = 0; i < 32; ++i) {
        const float4 v = row[i * 32 + lane];
        const int q0 = clampi((int)rintf(v.x * inv), -128, 127);
        const int q1 = clampi((int)rintf(v.y * inv), -128, 127);
        const int q2 = clampi((int)rintf(v.z * inv), -128, 127);
        const int q3 = clampi((int)rintf(v.w * inv), -128, 127);
        const uint32_t p = (uint32_t)(q0 & 0xff)         |
                           ((uint32_t)(q1 & 0xff) << 8)  |
                           ((uint32_t)(q2 & 0xff) << 16) |
                           ((uint32_t)(q3 & 0xff) << 24);
        qx[(size_t)token * (IN_F / 4) + i * 32 + lane] = p;
    }
}

__global__ __launch_bounds__(256) void Int8Linear_gemm(const uint8_t* __restrict__ qx,
                                                       const uint8_t* __restrict__ w8,
                                                       const float* __restrict__ xs,
                                                       const float* __restrict__ wscale,
                                                       const float* __restrict__ bias,
                                                       float* __restrict__ out) {
    __shared__ uint8_t As[NBUF * TILE_B];
    __shared__ uint8_t Bs[NBUF * TILE_B];

    const int tid   = threadIdx.x;
    const int lane  = tid & 31;
    const int wave  = tid >> 5;
    const int waveM = wave & 1;
    const int waveN = wave >> 1;
    const int half  = lane >> 4;
    const int l16   = lane & 15;

    const int m0 = blockIdx.x * BM;
    const int n0 = blockIdx.y * BN;

    v8i acc[4][2] = {};

    uint64_t gA[2], gB[2];
    uint32_t ldsAo[2], ldsBo[2];
#pragma unroll
    for (int c = 0; c < 2; ++c) {
        const int idx = tid + c * 256;
        const int row = idx >> 2;
        const int col = (idx & 3) * 16;
        gA[c] = (uint64_t)(size_t)(qx + (size_t)(m0 + row) * IN_F + col);
        gB[c] = (uint64_t)(size_t)(w8 + (size_t)(n0 + row) * IN_F + col);
        ldsAo[c] = (uint32_t)(size_t)(&As[0]) + (uint32_t)(row * LDT + col);
        ldsBo[c] = (uint32_t)(size_t)(&Bs[0]) + (uint32_t)(row * LDT + col);
    }

    auto issueTile = [&](int buf) {
#pragma unroll
        for (int c = 0; c < 2; ++c) {
            const uint32_t la = ldsAo[c] + (uint32_t)(buf * TILE_B);
            const uint32_t lb = ldsBo[c] + (uint32_t)(buf * TILE_B);
            asm volatile("global_load_async_to_lds_b128 %0, %1, off"
                         :: "v"(la), "v"(gA[c]) : "memory");
            asm volatile("global_load_async_to_lds_b128 %0, %1, off"
                         :: "v"(lb), "v"(gB[c]) : "memory");
            gA[c] += BK;
            gB[c] += BK;
        }
    };

    issueTile(0);
    issueTile(1);
    issueTile(2);

    for (int kt = 0; kt < NT; ++kt) {
        const int buf = kt & (NBUF - 1);

        asm volatile("s_wait_asynccnt 8" ::: "memory");
        __syncthreads();
        issueTile((kt + 3) & (NBUF - 1));

        const uint8_t* Ab = &As[buf * TILE_B];
        const uint8_t* Bb = &Bs[buf * TILE_B];

        v8i afrag[4];
#pragma unroll
        for (int mt = 0; mt < 4; ++mt) {
            const uint8_t* base = Ab + (waveM * 64 + mt * 16 + l16) * LDT;
            const uint4 d0 = *(const uint4*)(base + half * 16);
            const uint4 d1 = *(const uint4*)(base + 32 + half * 16);
            v8i a;
            a[0] = (int)d0.x; a[1] = (int)d0.y;
            a[2] = (int)d0.z; a[3] = (int)d0.w;
            a[4] = (int)d1.x; a[5] = (int)d1.y;
            a[6] = (int)d1.z; a[7] = (int)d1.w;
            afrag[mt] = a;
        }
        v8i bfrag[2];
#pragma unroll
        for (int nt = 0; nt < 2; ++nt) {
            const uint8_t* base = Bb + (waveN * 32 + nt * 16 + l16) * LDT;
            const uint4 lo = *(const uint4*)(base + half * 16);
            const uint4 hi = *(const uint4*)(base + 32 + half * 16);
            v8i b;
            b[0] = (int)lo.x; b[1] = (int)lo.y; b[2] = (int)lo.z; b[3] = (int)lo.w;
            b[4] = (int)hi.x; b[5] = (int)hi.y; b[6] = (int)hi.z; b[7] = (int)hi.w;
            bfrag[nt] = b;
        }

#pragma unroll
        for (int mt = 0; mt < 4; ++mt)
#pragma unroll
            for (int nt = 0; nt < 2; ++nt)
                acc[mt][nt] = __builtin_amdgcn_wmma_i32_16x16x64_iu8(
                     true, afrag[mt],
                     true, bfrag[nt],
                    acc[mt][nt],  false,  false);
    }

#pragma unroll
    for (int mt = 0; mt < 4; ++mt) {
#pragma unroll
        for (int nt = 0; nt < 2; ++nt) {
            const int n = n0 + waveN * 32 + nt * 16 + l16;
            const float ws = wscale[n];
            const _Float16 bi = (_Float16)bias[n];
#pragma unroll
            for (int v = 0; v < 8; ++v) {
                const int m = m0 + waveM * 64 + mt * 16 + v + half * 8;
                const float val = (float)acc[mt][nt][v] * ws * xs[m];
                _Float16 h = (_Float16)val;
                h = h + bi;
                out[(size_t)m * OUT_F + n] = (float)h;
            }
        }
    }
}

extern "C" void kernel_launch(void* const* d_in, const int* in_sizes, int n_in,
                              void* d_out, int out_size, void* d_ws, size_t ws_size,
                              hipStream_t stream) {
    const float* x      = (const float*)d_in[0];
    const int*   w      = (const int*)d_in[1];
    const float* wscale = (const float*)d_in[2];
    const float* bias   = (const float*)d_in[3];
    float* out = (float*)d_out;

    const int T = in_sizes[0] / IN_F;

    uint8_t* ws8 = (uint8_t*)d_ws;
    uint32_t* w8 = (uint32_t*)ws8;
    const size_t w8_bytes = (size_t)OUT_F * IN_F;
    uint32_t* qx = (uint32_t*)(ws8 + w8_bytes);
    const size_t qx_bytes = (size_t)T * IN_F;
    float* xs = (float*)(ws8 + w8_bytes + qx_bytes);

    Int8Linear_repack_w<<<(OUT_F * IN_F / 4) / 256, 256, 0, stream>>>(w, w8);
    Int8Linear_quantize<<<T / 8, 256, 0, stream>>>(x, qx, xs);

    dim3 grid(T / BM, OUT_F / BN);
    Int8Linear_gemm<<<grid, 256, 0, stream>>>((const uint8_t*)qx, (const uint8_t*)w8,
                                              xs, wscale, bias, out);
}
